// DualRelGCN_14482629722333
// MI455X (gfx1250) — matrix operand order repaired and checked by reading, not compiled, not run on hardware
//
#include <hip/hip_runtime.h>
#include <hip/hip_bf16.h>

#define NNODES 10000
#define DIM 256
#define ALPHA_F 0.1f
#define EPS_DENOM 1e-12f
#define LN_EPS 1e-5f

typedef __attribute__((ext_vector_type(16))) __bf16 v16bf;
typedef __attribute__((ext_vector_type(8)))  float  v8f;

__device__ __forceinline__ unsigned short f32_to_bf16_rne(float f) {
    unsigned u = __builtin_bit_cast(unsigned, f);
    u += 0x7FFFu + ((u >> 16) & 1u);   // round-to-nearest-even
    return (unsigned short)(u >> 16);
}

// ---------------- zero scratch (agg + denom are contiguous) ----------------
__global__ void zero_f32(float* __restrict__ p, int n) {
    int i = blockIdx.x * blockDim.x + threadIdx.x;
    if (i < n) p[i] = 0.0f;
}

// ---------------- proj_w (f32, row-major [DIM][DIM]) -> bf16 bits ----------
__global__ void convert_w_bf16(const float* __restrict__ w,
                               unsigned short* __restrict__ wbf, int n) {
    int i = blockIdx.x * blockDim.x + threadIdx.x;
    if (i < n) wbf[i] = f32_to_bf16_rne(w[i]);
}

// ---------------- edge scatter: one wave32 per edge ------------------------
__global__ void edge_scatter(const float* __restrict__ x,
                             const long long* __restrict__ idx,
                             const float* __restrict__ ew,
                             float* __restrict__ agg,
                             float* __restrict__ denom,
                             int E) {
    int gwave = (blockIdx.x * blockDim.x + threadIdx.x) >> 5;
    int lane  = threadIdx.x & 31;
    int nwave = (gridDim.x * blockDim.x) >> 5;
    for (int e = gwave; e < E; e += nwave) {
        long long s = idx[e];
        long long d = idx[E + e];
        float wt = ew[e];
        const float4* xr = (const float4*)(x + (size_t)s * DIM);
        float4 a = xr[lane];        // cols [4*lane, 4*lane+3]
        float4 b = xr[lane + 32];   // cols [128+4*lane, ...]
        float* ar = agg + (size_t)d * DIM;
        unsafeAtomicAdd(ar + 4 * lane + 0,       a.x * wt);
        unsafeAtomicAdd(ar + 4 * lane + 1,       a.y * wt);
        unsafeAtomicAdd(ar + 4 * lane + 2,       a.z * wt);
        unsafeAtomicAdd(ar + 4 * lane + 3,       a.w * wt);
        unsafeAtomicAdd(ar + 128 + 4 * lane + 0, b.x * wt);
        unsafeAtomicAdd(ar + 128 + 4 * lane + 1, b.y * wt);
        unsafeAtomicAdd(ar + 128 + 4 * lane + 2, b.z * wt);
        unsafeAtomicAdd(ar + 128 + 4 * lane + 3, b.w * wt);
        if (lane == 0) unsafeAtomicAdd(denom + d, wt);
    }
}

// ---------------- fused normalize + LayerNorm + GEMM(WMMA) + residual ------
// One block = 16 node rows. 8 waves; wave w owns output col tiles 2w, 2w+1.
__global__ void __launch_bounds__(256)
fused_ln_gemm(const float* __restrict__ agg,
              const float* __restrict__ denom,
              const unsigned short* __restrict__ wbf,
              const float* __restrict__ x,
              float* __restrict__ out) {
    // padded stride 264 -> A-fragment rows land on distinct LDS banks
    __shared__ __align__(16) unsigned short sA[16][DIM + 8];

    const int tid  = threadIdx.x;
    const int wave = tid >> 5;
    const int lane = tid & 31;
    const int g    = lane >> 4;   // half-wave group (selects K offset)
    const int l16  = lane & 15;
    const int rowBase = blockIdx.x * 16;

    // ---- Phase 1: LayerNorm(agg/denom) -> bf16 tile in LDS ----
    {
        int row = 2 * wave + g;                  // 0..15, all rows covered
        int rg  = rowBase + row;
        float inv = 1.0f / fmaxf(denom[rg], EPS_DENOM);
        const float4* ap = (const float4*)(agg + (size_t)rg * DIM + l16 * 16);
        float4 v0 = ap[0], v1 = ap[1], v2 = ap[2], v3 = ap[3];
        float vals[16] = { v0.x, v0.y, v0.z, v0.w, v1.x, v1.y, v1.z, v1.w,
                           v2.x, v2.y, v2.z, v2.w, v3.x, v3.y, v3.z, v3.w };
        float s = 0.0f, s2 = 0.0f;
        #pragma unroll
        for (int i = 0; i < 16; ++i) {
            vals[i] *= inv;
            s  += vals[i];
            s2 += vals[i] * vals[i];
        }
        // reduce across the 16 lanes that share this row
        #pragma unroll
        for (int m = 1; m < 16; m <<= 1) {
            s  += __shfl_xor(s,  m, 32);
            s2 += __shfl_xor(s2, m, 32);
        }
        float mean = s * (1.0f / DIM);
        float var  = s2 * (1.0f / DIM) - mean * mean;
        float rstd = rsqrtf(var + LN_EPS);
        #pragma unroll
        for (int i = 0; i < 16; ++i)
            sA[row][l16 * 16 + i] = f32_to_bf16_rne((vals[i] - mean) * rstd);
    }
    __syncthreads();

    // ---- Phase 2: D = A(16x256) x W^T tiles via v_wmma_f32_16x16x32_bf16 ----
    v8f acc0 = {};
    v8f acc1 = {};
    const int c0 = 2 * wave, c1 = 2 * wave + 1;
    const int rw0 = (c0 * 16 + l16) * DIM;  // W row for output col c0*16+l16
    const int rw1 = (c1 * 16 + l16) * DIM;

    #pragma unroll
    for (int kt = 0; kt < DIM / 32; ++kt) {
        // A fragment: lane row M=l16; element h -> K = h + 16g (contiguous)
        union { unsigned u[8]; v16bf v; } A;
        const int abase = kt * 32 + 16 * g;
        #pragma unroll
        for (int i = 0; i < 8; ++i) {
            int kidx = abase + 2 * i;
            A.u[i] = *(const unsigned*)&sA[l16][kidx];
        }
        // B fragment: lane col N=l16; element h -> K = h + 16g (contiguous)
        union { uint4 q[2]; v16bf v; } B0, B1;
        const uint4* pb0 = (const uint4*)(wbf + rw0 + kt * 32 + 16 * g);
        B0.q[0] = pb0[0]; B0.q[1] = pb0[1];
        const uint4* pb1 = (const uint4*)(wbf + rw1 + kt * 32 + 16 * g);
        B1.q[0] = pb1[0]; B1.q[1] = pb1[1];

        acc0 = __builtin_amdgcn_wmma_f32_16x16x32_bf16(
            false, A.v, false, B0.v, (short)0, acc0, false, false);
        acc1 = __builtin_amdgcn_wmma_f32_16x16x32_bf16(
            false, A.v, false, B1.v, (short)0, acc1, false, false);
    }

    // ---- Epilogue: out = rel_embed + ALPHA * delta ----
    #pragma unroll
    for (int r = 0; r < 8; ++r) {
        int m = r + 8 * g;                       // D row in this lane group
        size_t o0 = (size_t)(rowBase + m) * DIM + c0 * 16 + l16;
        out[o0] = x[o0] + ALPHA_F * acc0[r];
        size_t o1 = (size_t)(rowBase + m) * DIM + c1 * 16 + l16;
        out[o1] = x[o1] + ALPHA_F * acc1[r];
    }
}

extern "C" void kernel_launch(void* const* d_in, const int* in_sizes, int n_in,
                              void* d_out, int out_size, void* d_ws, size_t ws_size,
                              hipStream_t stream) {
    const float*     x   = (const float*)d_in[0];      // rel_embed  [N, DIM] f32
    const long long* idx = (const long long*)d_in[1];  // edge_index [2, E]  i64
    const float*     ew  = (const float*)d_in[2];      // edge_weight [E]    f32
    const float*     pw  = (const float*)d_in[3];      // proj_w [DIM, DIM]  f32
    float*           out = (float*)d_out;
    const int E = in_sizes[2];

    // workspace layout: agg[N*DIM] f32 | denom[N] f32 | wbf[DIM*DIM] bf16-bits
    float* agg   = (float*)d_ws;
    float* denom = agg + (size_t)NNODES * DIM;
    unsigned short* wbf = (unsigned short*)(denom + NNODES);

    const int nz = NNODES * DIM + NNODES;
    zero_f32<<<(nz + 255) / 256, 256, 0, stream>>>(agg, nz);
    convert_w_bf16<<<(DIM * DIM + 255) / 256, 256, 0, stream>>>(pw, wbf, DIM * DIM);
    edge_scatter<<<2048, 256, 0, stream>>>(x, idx, ew, agg, denom, E);
    fused_ln_gemm<<<NNODES / 16, 256, 0, stream>>>(agg, denom, wbf, x, out);
}
